// AttentionBlock_38989713113205
// MI455X (gfx1250) — matrix operand order repaired and checked by reading, not compiled, not run on hardware
//
#include <hip/hip_runtime.h>
#include <hip/hip_bf16.h>

typedef __attribute__((ext_vector_type(16))) _Float16 v16h;
typedef __attribute__((ext_vector_type(8)))  _Float16 v8h;
typedef __attribute__((ext_vector_type(4)))  _Float16 v4h;
typedef __attribute__((ext_vector_type(8)))  float    v8f;

#define HW     4096
#define C_IN   256
#define O_ALL  320     // 32 q + 32 k + 256 v output channels
#define MTILE  128     // pixels per block
#define KTILE  32      // K-depth per WMMA chunk
#define NCHUNK (C_IN / KTILE)
#define LDSPAD 40      // row pitch in halves: 80B, 16B-aligned, bank-friendly

// ---------------------------------------------------------------------------
// Kernel 1: fused q/k/v projection GEMM.  Y[b, o, m] = W_all[o,:]·x[b,:,m] + bias
// Block: 512 threads = 16 waves; one batch, 128 pixels per block.
// Software-pipelined: chunk kc+1 is prefetched into registers (float4 vmem)
// while chunk kc's 10 WMMAs run out of LDS.
// ---------------------------------------------------------------------------
__global__ __launch_bounds__(512)
void proj_kernel(const float* __restrict__ x,
                 const float* __restrict__ Wq, const float* __restrict__ bq,
                 const float* __restrict__ Wk, const float* __restrict__ bk,
                 const float* __restrict__ Wv, const float* __restrict__ bvp,
                 float* __restrict__ Y)
{
    __shared__ __align__(16) _Float16 wsW[O_ALL][LDSPAD];  // 25.6 KB
    __shared__ __align__(16) _Float16 xs[MTILE][LDSPAD];   // 10.2 KB
    __shared__ float bias_s[O_ALL];

    const int tid = threadIdx.x;
    const int b   = blockIdx.x >> 5;          // batch
    const int m0  = (blockIdx.x & 31) * MTILE;

    if (tid < O_ALL) {
        float bb;
        if      (tid < 32) bb = bq[tid];
        else if (tid < 64) bb = bk[tid - 32];
        else               bb = bvp[tid - 64];
        bias_s[tid] = bb;
    }

    const int lane = tid & 31;
    const int wv   = tid >> 5;          // 0..15
    const int pt   = wv & 7;            // pixel tile within block
    const int og   = wv >> 3;           // 0..1: which 10 o-tiles
    const int mrow = pt * 16 + (lane & 15);   // xs row for B fragment
    const int bKb  = (lane >> 4) * 8;         // B: lanes 16-31 hold K+8 runs
    const int aKb  = (lane >> 4) * 8;         // A: lanes 16-31 hold K+8 runs

    // W tile: 320x32 = 2560 float4, 5 per thread.  X tile: 32x128 = 1024 float4, 2/thread.
    float4 wreg[5], xreg[2];

    auto load_chunk = [&](int c0) {
        #pragma unroll
        for (int i = 0; i < 5; ++i) {
            int f = (tid + i * 512) * 4;          // flat float index in 320x32 tile
            int o = f >> 5, cl = f & 31;          // cl is a multiple of 4
            const float* src;
            if      (o < 32) src = Wq + (size_t)o * C_IN;
            else if (o < 64) src = Wk + (size_t)(o - 32) * C_IN;
            else             src = Wv + (size_t)(o - 64) * C_IN;
            wreg[i] = *(const float4*)(src + c0 + cl);
        }
        #pragma unroll
        for (int i = 0; i < 2; ++i) {
            int f  = (tid + i * 512) * 4;         // flat float index in 32x128 tile
            int kl = f >> 7, ml = f & 127;        // ml is a multiple of 4
            xreg[i] = *(const float4*)(x + ((size_t)b * C_IN + c0 + kl) * HW + m0 + ml);
        }
    };

    auto store_chunk = [&]() {
        #pragma unroll
        for (int i = 0; i < 5; ++i) {
            int f = (tid + i * 512) * 4;
            int o = f >> 5, cl = f & 31;
            v4h h;
            h[0] = (_Float16)wreg[i].x; h[1] = (_Float16)wreg[i].y;
            h[2] = (_Float16)wreg[i].z; h[3] = (_Float16)wreg[i].w;
            *(v4h*)&wsW[o][cl] = h;               // ds_store_b64
        }
        #pragma unroll
        for (int i = 0; i < 2; ++i) {
            int f  = (tid + i * 512) * 4;
            int kl = f >> 7, ml = f & 127;
            xs[ml + 0][kl] = (_Float16)xreg[i].x; // transpose to [pixel][k]
            xs[ml + 1][kl] = (_Float16)xreg[i].y;
            xs[ml + 2][kl] = (_Float16)xreg[i].z;
            xs[ml + 3][kl] = (_Float16)xreg[i].w;
        }
    };

    v8f acc[10] = {};

    load_chunk(0);                                // prologue prefetch

    for (int kc = 0; kc < NCHUNK; ++kc) {
        __syncthreads();     // previous iteration's LDS reads complete
        store_chunk();
        __syncthreads();

        if (kc + 1 < NCHUNK) load_chunk((kc + 1) * KTILE);  // overlap with WMMAs

        // B fragment: lane holds pixel N=lane&15, halves 0-7 -> K=bKb..bKb+7, 8-15 -> K=bKb+16..bKb+23
        v8h blo = *(const v8h*)&xs[mrow][bKb];
        v8h bhi = *(const v8h*)&xs[mrow][bKb + 16];
        v16h bfrag = __builtin_shufflevector(blo, bhi,
            0,1,2,3,4,5,6,7,8,9,10,11,12,13,14,15);

        #pragma unroll
        for (int t = 0; t < 10; ++t) {
            // A fragment: lane row M=lane&15; halves 0-7 -> K=aKb..aKb+7,
            //             halves 8-15 -> K=aKb+16..aKb+23
            int orow = (og * 10 + t) * 16 + (lane & 15);
            v8h alo = *(const v8h*)&wsW[orow][aKb];
            v8h ahi = *(const v8h*)&wsW[orow][aKb + 16];
            v16h afrag = __builtin_shufflevector(alo, ahi,
                0,1,2,3,4,5,6,7,8,9,10,11,12,13,14,15);
            acc[t] = __builtin_amdgcn_wmma_f32_16x16x32_f16(
                false, afrag, false, bfrag, (short)0, acc[t], false, false);
        }
    }

    // C/D layout: VGPR r -> M = r + 8*(lane>=16), N = lane&15
    const int m = m0 + pt * 16 + (lane & 15);
    #pragma unroll
    for (int t = 0; t < 10; ++t) {
        #pragma unroll
        for (int r = 0; r < 8; ++r) {
            int o = (og * 10 + t) * 16 + r + ((lane >> 4) << 3);
            Y[((size_t)b * O_ALL + o) * HW + m] = acc[t][r] + bias_s[o];
        }
    }
}

// ---------------------------------------------------------------------------
// Kernel 2: 3x3-window attention + residual. One thread per pixel; lanes are
// consecutive x-coords -> coalesced channel-strided loads. Zero-padding
// semantics: OOB neighbors have energy 0 (included in softmax) and value 0.
// ---------------------------------------------------------------------------
__global__ __launch_bounds__(256)
void attn_kernel(const float* __restrict__ Y, const float* __restrict__ x,
                 const float* __restrict__ gamma, float* __restrict__ out)
{
    const int t  = blockIdx.x * 256 + threadIdx.x;   // 0..32767
    const int b  = t >> 12;
    const int p  = t & 4095;
    const int yy = p >> 6;
    const int xx = p & 63;

    const float* q = Y + (size_t)b * O_ALL * HW;
    const float* k = q + 32 * HW;
    const float* v = q + 64 * HW;

    float qv[32];
    #pragma unroll
    for (int c = 0; c < 32; ++c) qv[c] = q[c * HW + p];

    float e[9], msk[9];
    int   np[9];
    #pragma unroll
    for (int j = 0; j < 9; ++j) {
        int dy = j / 3 - 1, dx = j % 3 - 1;
        int ny = yy + dy, nx = xx + dx;
        bool ib = ((unsigned)ny < 64u) && ((unsigned)nx < 64u);
        int  nq = ib ? (ny * 64 + nx) : p;    // safe index when OOB (weight 0)
        np[j]  = nq;
        msk[j] = ib ? 1.f : 0.f;
        float s = 0.f;
        if (ib) {
            for (int c = 0; c < 32; ++c) s = fmaf(qv[c], k[c * HW + nq], s);
        }
        e[j] = ib ? s : 0.f;                  // zero-pad => energy 0, kept in softmax
    }

    float mx = e[0];
    #pragma unroll
    for (int j = 1; j < 9; ++j) mx = fmaxf(mx, e[j]);
    float sum = 0.f;
    #pragma unroll
    for (int j = 0; j < 9; ++j) { e[j] = __expf(e[j] - mx); sum += e[j]; }
    const float inv = 1.f / sum;
    #pragma unroll
    for (int j = 0; j < 9; ++j) e[j] *= inv * msk[j];  // OOB values get weight 0

    const float g = gamma[0];
    for (int c = 0; c < 256; ++c) {
        float accv = 0.f;
        #pragma unroll
        for (int j = 0; j < 9; ++j) accv = fmaf(e[j], v[c * HW + np[j]], accv);
        size_t oidx = ((size_t)b * 256 + c) * HW + p;
        out[oidx] = fmaf(g, accv, x[oidx]);
    }
}

// ---------------------------------------------------------------------------
extern "C" void kernel_launch(void* const* d_in, const int* in_sizes, int n_in,
                              void* d_out, int out_size, void* d_ws, size_t ws_size,
                              hipStream_t stream)
{
    const float* x   = (const float*)d_in[0];
    const float* Wq  = (const float*)d_in[1];
    const float* bq  = (const float*)d_in[2];
    const float* Wk  = (const float*)d_in[3];
    const float* bk  = (const float*)d_in[4];
    const float* Wv  = (const float*)d_in[5];
    const float* bv  = (const float*)d_in[6];
    const float* gm  = (const float*)d_in[7];
    float* Y   = (float*)d_ws;                // 8*320*4096 fp32 = 41.9 MB
    float* out = (float*)d_out;

    hipLaunchKernelGGL(proj_kernel, dim3(8 * 32), dim3(512), 0, stream,
                       x, Wq, bq, Wk, bk, Wv, bv, Y);
    hipLaunchKernelGGL(attn_kernel, dim3(128), dim3(256), 0, stream,
                       Y, x, gm, out);
}
